// CorrelationOnlyTracker_19628000543429
// MI455X (gfx1250) — matrix operand order repaired and checked by reading, not compiled, not run on hardware
//
#include <hip/hip_runtime.h>
#include <hip/hip_bf16.h>
#include <stdint.h>

typedef __attribute__((ext_vector_type(16))) _Float16 v16h;
typedef __attribute__((ext_vector_type(8)))  _Float16 v8h;
typedef __attribute__((ext_vector_type(8)))  float    v8f;

// Problem constants
#define BB 2
#define TT 24
#define HH 256
#define WW 256
#define NQ 512
#define CH 256          // HIDDEN_DIM
#define FH 32
#define FW 32
#define FRAMES (BB*TT)

// ---------------------------------------------------------------------------
// Utility kernels
// ---------------------------------------------------------------------------
__global__ void cast_video_kernel(const float* __restrict__ src,
                                  _Float16* __restrict__ dst, int n) {
    int i = blockIdx.x * blockDim.x + threadIdx.x;
    if (i < n) dst[i] = (_Float16)(src[i] * (1.0f / 255.0f));
}

// Pack conv weights [ktot][cout] (k = (ky*KW+kx)*CIN+ci, row-major in src)
// straight into WMMA B-fragment order: dst[((kc*OCT + oct)*32 + lane)*16 + e]
// where lane<16 holds K=kc*32+0..15 of column oc, lane>=16 holds K=+16..31.
// Rows k >= ktot are zero (K padding).
__global__ void pack_wmma_b_kernel(const float* __restrict__ src,
                                   _Float16* __restrict__ dst,
                                   int ktot, int kpad, int cout) {
    int i = blockIdx.x * blockDim.x + threadIdx.x;
    int total = kpad * cout;
    if (i >= total) return;
    int e    = i & 15;
    int l    = (i >> 4) & 31;
    int rest = i >> 9;                 // kc*OCT + oct
    int octn = cout >> 4;
    int oct  = rest % octn;
    int kc   = rest / octn;
    int half = l >> 4;
    int oc   = oct * 16 + (l & 15);
    int k    = kc * 32 + half * 16 + e;
    _Float16 v = (_Float16)0.0f;
    if (k < ktot) v = (_Float16)src[(size_t)k * cout + oc];
    dst[i] = v;
}

// ---------------------------------------------------------------------------
// Generic WMMA conv: one wave computes 16 output pixels x 16 output channels.
// Input  : f16 [F][HIN][WIN][CIN]
// Weights: f16 fragment-packed [KPAD/32][COUT/16][32][16]
// Output : f16 [F][HOUT][WOUT][COUT], ReLU(out + bias)
// ---------------------------------------------------------------------------
template<int KH,int KW,int CIN,int STRIDE,int PAD,int HIN,int WIN>
__device__ inline void load_a_group8(const _Float16* __restrict__ in,
                                     int f, int oy, int ox, int ks,
                                     _Float16* dst, bool prefetch_next) {
    constexpr int KTOT = KH * KW * CIN;
    if constexpr (CIN % 32 == 0) {
        // ks is a multiple of 8; the 8-run stays within one (ky,kx) cell.
        int ky = ks / (KW * CIN);
        int rem = ks % (KW * CIN);
        int kx = rem / CIN;
        int ci = rem % CIN;
        int iy = oy * STRIDE - PAD + ky;
        int ix = ox * STRIDE - PAD + kx;
        if (iy >= 0 && iy < HIN && ix >= 0 && ix < WIN) {
            const _Float16* base =
                in + (((size_t)f * HIN + iy) * WIN + ix) * CIN + ci;
            v8h g = *reinterpret_cast<const v8h*>(base);
#pragma unroll
            for (int j = 0; j < 8; ++j) dst[j] = g[j];
            if (prefetch_next) __builtin_prefetch(base + 32, 0, 1);
        } else {
#pragma unroll
            for (int j = 0; j < 8; ++j) dst[j] = (_Float16)0.0f;
        }
    } else {
#pragma unroll
        for (int j = 0; j < 8; ++j) {
            int k = ks + j;
            _Float16 v = (_Float16)0.0f;
            if (k < KTOT) {
                int ky = k / (KW * CIN);
                int rem = k % (KW * CIN);
                int kx = rem / CIN;
                int ci = rem % CIN;
                int iy = oy * STRIDE - PAD + ky;
                int ix = ox * STRIDE - PAD + kx;
                if (iy >= 0 && iy < HIN && ix >= 0 && ix < WIN)
                    v = in[(((size_t)f * HIN + iy) * WIN + ix) * CIN + ci];
            }
            dst[j] = v;
        }
    }
}

template<int KH,int KW,int CIN,int COUT,int STRIDE,int PAD,
         int HIN,int WIN,int HOUT,int WOUT,int KPAD>
__global__ __launch_bounds__(256)
void conv_wmma_kernel(const _Float16* __restrict__ in,
                      const _Float16* __restrict__ wgt,
                      const float*    __restrict__ bias,
                      _Float16*       __restrict__ out,
                      int frames) {
    constexpr int XT  = WOUT / 16;
    constexpr int OCT = COUT / 16;
    int wave = blockIdx.x * (blockDim.x >> 5) + (threadIdx.x >> 5);
    int oct = wave % OCT;  wave /= OCT;
    int xt  = wave % XT;   wave /= XT;
    int oy  = wave % HOUT; wave /= HOUT;
    int f   = wave;
    if (f >= frames) return;

    int lane  = threadIdx.x & 31;
    int half  = lane >> 4;   // which K-half this lane holds
    int idx16 = lane & 15;

    int ox = xt * 16 + idx16;     // A-matrix row (pixel) for this lane
    int oc = oct * 16 + idx16;    // B-matrix column (out channel)

    // Per-lane fragment-packed weight pointer for this oct tile.
    const _Float16* wlane = wgt + (((size_t)oct * 32 + lane) << 4);
    constexpr size_t WSTEP = (size_t)OCT * 32 * 16;   // halfs per K-chunk

    v8f acc = {};
    for (int kc = 0; kc < KPAD / 32; ++kc) {
        int kbase = kc * 32;
        // ---- A fragment: im2col row for pixel `ox`, contiguous K = 16*half + e, as in B ----
        _Float16 tmp[16];
        load_a_group8<KH,KW,CIN,STRIDE,PAD,HIN,WIN>(
            in, f, oy, ox, kbase + half * 16, tmp, kc + 1 < KPAD / 32);
        load_a_group8<KH,KW,CIN,STRIDE,PAD,HIN,WIN>(
            in, f, oy, ox, kbase + half * 16 + 8, tmp + 8, false);
        v16h a = {};
#pragma unroll
        for (int e = 0; e < 16; ++e) a[e] = tmp[e];

        // ---- B fragment: single contiguous 32B load (fragment-packed) ----
        v16h b = *reinterpret_cast<const v16h*>(wlane + (size_t)kc * WSTEP);

        acc = __builtin_amdgcn_wmma_f32_16x16x32_f16(
            false, a, false, b, (short)0, acc, false, false);
    }

    float bv = bias[oc];
#pragma unroll
    for (int r = 0; r < 8; ++r) {
        int row = r + 8 * half;           // D-matrix row = pixel in tile
        int ox_out = xt * 16 + row;
        float v = acc[r] + bv;
        v = v > 0.0f ? v : 0.0f;
        out[(((size_t)f * HOUT + oy) * WOUT + ox_out) * COUT + oc] = (_Float16)v;
    }
}

// ---------------------------------------------------------------------------
// Bilinear query feature sampling: q[b][n][c] from feat3
// ---------------------------------------------------------------------------
__global__ __launch_bounds__(256)
void sample_query_kernel(const _Float16* __restrict__ feat3,
                         const float*    __restrict__ qp,
                         _Float16*       __restrict__ qh) {
    int bn = blockIdx.x;          // b*NQ + n
    int c  = threadIdx.x;         // channel
    int b  = bn / NQ;
    const float* q = qp + (size_t)bn * 3;
    float qt = q[0] * (float)(TT - 1);
    float qy = q[1] * (float)(FH - 1);
    float qx = q[2] * (float)(FW - 1);
    int t = (int)qt; t = t < 0 ? 0 : (t > TT - 1 ? TT - 1 : t);
    int y0 = (int)floorf(qy); y0 = y0 < 0 ? 0 : (y0 > FH - 1 ? FH - 1 : y0);
    int y1 = y0 + 1 > FH - 1 ? FH - 1 : y0 + 1;
    int x0 = (int)floorf(qx); x0 = x0 < 0 ? 0 : (x0 > FW - 1 ? FW - 1 : x0);
    int x1 = x0 + 1 > FW - 1 ? FW - 1 : x0 + 1;
    float wy1 = qy - (float)y0, wy0 = 1.0f - wy1;
    float wx1 = qx - (float)x0, wx0 = 1.0f - wx1;

    int fidx = b * TT + t;
    const _Float16* base = feat3 + ((size_t)fidx * FH * FW) * CH + c;
    float f00 = (float)base[((size_t)y0 * FW + x0) * CH];
    float f01 = (float)base[((size_t)y0 * FW + x1) * CH];
    float f10 = (float)base[((size_t)y1 * FW + x0) * CH];
    float f11 = (float)base[((size_t)y1 * FW + x1) * CH];
    float v = (f00 * wx0 + f01 * wx1) * wy0 + (f10 * wx0 + f11 * wx1) * wy1;
    qh[(size_t)bn * CH + c] = (_Float16)v;
}

// ---------------------------------------------------------------------------
// Fused correlation (WMMA GEMM) + softmax + soft-argmax + occlusion.
// Block = (b, 16-query tile, t); 128 threads = 4 waves.
// ---------------------------------------------------------------------------
__global__ __launch_bounds__(128)
void corr_softmax_kernel(const _Float16* __restrict__ qh,
                         const _Float16* __restrict__ feat3,
                         float* __restrict__ out) {
    __shared__ _Float16 s_corr[16 * 1024];     // [query_m][pos], corr/16
    __shared__ float s_max[16][8];
    __shared__ float s_se[16][8];
    __shared__ float s_sx[16][8];
    __shared__ float s_sy[16][8];

    int id = blockIdx.x;
    int tf = id % TT;            id /= TT;
    int nt = id % (NQ / 16);     id /= (NQ / 16);
    int b  = id;
    int fidx = b * TT + tf;

    int wave  = threadIdx.x >> 5;
    int lane  = threadIdx.x & 31;
    int half  = lane >> 4;
    int idx16 = lane & 15;

    // Preload A fragments (16 queries x K=256) for this wave's lanes.
    int n = nt * 16 + idx16;     // A-matrix row = query
    const _Float16* qbase = qh + ((size_t)b * NQ + n) * CH;
    v16h afrag[8];
#pragma unroll
    for (int kc = 0; kc < 8; ++kc) {
        int ks0 = kc * 32 + half * 16;
        const v8h* p0 = reinterpret_cast<const v8h*>(qbase + ks0);
        const v8h* p1 = reinterpret_cast<const v8h*>(qbase + ks0 + 8);
        v8h g0 = *p0, g1 = *p1;
        v16h a = {};
#pragma unroll
        for (int j = 0; j < 8; ++j) { a[j] = g0[j]; a[j + 8] = g1[j]; }
        afrag[kc] = a;
    }

    // 64 position tiles split across 4 waves.
    for (int pt = wave; pt < 64; pt += 4) {
        int pos = pt * 16 + idx16;   // B-matrix column = position
        v8f acc = {};
        const _Float16* fbase = feat3 + ((size_t)fidx * (FH * FW) + pos) * CH;
        if (pt + 4 < 64) __builtin_prefetch(fbase + 64 * CH, 0, 1);
        for (int kc = 0; kc < 8; ++kc) {
            const v16h* bp = reinterpret_cast<const v16h*>(fbase + kc * 32 + half * 16);
            v16h bfr = *bp;
            acc = __builtin_amdgcn_wmma_f32_16x16x32_f16(
                false, afrag[kc], false, bfr, (short)0, acc, false, false);
        }
#pragma unroll
        for (int r = 0; r < 8; ++r) {
            int m = r + 8 * half;                 // query row within tile
            s_corr[m * 1024 + pt * 16 + idx16] = (_Float16)(acc[r] * 0.0625f);
        }
    }
    __syncthreads();

    // Reduction: 8 threads per query row.
    int m    = threadIdx.x >> 3;      // 0..15
    int part = threadIdx.x & 7;       // 0..7

    float lmax = -3.4e38f;
    for (int p = part; p < 1024; p += 8) {
        float v = (float)s_corr[m * 1024 + p];
        lmax = v > lmax ? v : lmax;
    }
    s_max[m][part] = lmax;
    __syncthreads();
    float gmax = s_max[m][0];
#pragma unroll
    for (int j = 1; j < 8; ++j) gmax = s_max[m][j] > gmax ? s_max[m][j] : gmax;

    float se = 0.0f, sx = 0.0f, sy = 0.0f;
    for (int p = part; p < 1024; p += 8) {
        float v = (float)s_corr[m * 1024 + p];
        float e = __expf(10.0f * (v - gmax));
        se += e;
        sx += e * (float)(p & 31);
        sy += e * (float)(p >> 5);
    }
    s_se[m][part] = se;
    s_sx[m][part] = sx;
    s_sy[m][part] = sy;
    __syncthreads();

    if (part == 0) {
        float Se = 0.0f, Sx = 0.0f, Sy = 0.0f;
#pragma unroll
        for (int j = 0; j < 8; ++j) { Se += s_se[m][j]; Sx += s_sx[m][j]; Sy += s_sy[m][j]; }
        int nq = nt * 16 + m;
        size_t pidx = (((size_t)b * NQ + nq) * TT + tf) * 2;
        out[pidx]     = (Sx / Se) * (float)(WW / FW);   // x * 8
        out[pidx + 1] = (Sy / Se) * (float)(HH / FH);   // y * 8
        size_t occ_off = (size_t)BB * NQ * TT * 2;
        out[occ_off + ((size_t)b * NQ + nq) * TT + tf] = 1.0f / (1.0f + __expf(gmax));
    }
}

// ---------------------------------------------------------------------------
// Host launcher
// ---------------------------------------------------------------------------
extern "C" void kernel_launch(void* const* d_in, const int* in_sizes, int n_in,
                              void* d_out, int out_size, void* d_ws, size_t ws_size,
                              hipStream_t stream) {
    const float* video = (const float*)d_in[0];
    const float* qp    = (const float*)d_in[1];
    const float* w1    = (const float*)d_in[2];
    const float* b1    = (const float*)d_in[3];
    const float* w2    = (const float*)d_in[4];
    const float* b2    = (const float*)d_in[5];
    const float* w3    = (const float*)d_in[6];
    const float* b3    = (const float*)d_in[7];
    float* out = (float*)d_out;

    char* ws = (char*)d_ws;
    size_t off = 0;
    auto take = [&](size_t bytes) -> char* {
        char* p = ws + off;
        off = (off + bytes + 255) & ~(size_t)255;
        return p;
    };

    const size_t n_vid = (size_t)BB * TT * HH * WW * 3;
    _Float16* vh    = (_Float16*)take(n_vid * 2);
    _Float16* feat1 = (_Float16*)take((size_t)FRAMES * 128 * 128 * 64  * 2);
    _Float16* feat2 = (_Float16*)take((size_t)FRAMES * 64  * 64  * 128 * 2);
    _Float16* feat3 = (_Float16*)take((size_t)FRAMES * 32  * 32  * 256 * 2);
    _Float16* qh    = (_Float16*)take((size_t)BB * NQ * CH * 2);
    _Float16* w1h   = (_Float16*)take((size_t)160  * 64  * 2);
    _Float16* w2h   = (_Float16*)take((size_t)576  * 128 * 2);
    _Float16* w3h   = (_Float16*)take((size_t)1152 * 256 * 2);

    // 1. cast video
    {
        int n = (int)n_vid;
        cast_video_kernel<<<(n + 255) / 256, 256, 0, stream>>>(video, vh, n);
    }
    // 2. pack weights into WMMA B-fragment order
    pack_wmma_b_kernel<<<(160 * 64 + 255) / 256, 256, 0, stream>>>(w1, w1h, 147, 160, 64);
    pack_wmma_b_kernel<<<(576 * 128 + 255) / 256, 256, 0, stream>>>(w2, w2h, 576, 576, 128);
    pack_wmma_b_kernel<<<(1152 * 256 + 255) / 256, 256, 0, stream>>>(w3, w3h, 1152, 1152, 256);

    // 3. convs (WMMA)
    {   // conv1: 7x7, 3->64, stride 2, pad_lo 2 ; 196608 waves
        int waves = FRAMES * 128 * (128 / 16) * (64 / 16);
        conv_wmma_kernel<7,7,3,64,2,2, 256,256,128,128, 160>
            <<<waves / 8, 256, 0, stream>>>(vh, w1h, b1, feat1, FRAMES);
    }
    {   // conv2: 3x3, 64->128, stride 2, pad_lo 0 ; 98304 waves
        int waves = FRAMES * 64 * (64 / 16) * (128 / 16);
        conv_wmma_kernel<3,3,64,128,2,0, 128,128,64,64, 576>
            <<<waves / 8, 256, 0, stream>>>(feat1, w2h, b2, feat2, FRAMES);
    }
    {   // conv3: 3x3, 128->256, stride 2, pad_lo 0 ; 49152 waves
        int waves = FRAMES * 32 * (32 / 16) * (256 / 16);
        conv_wmma_kernel<3,3,128,256,2,0, 64,64,32,32, 1152>
            <<<waves / 8, 256, 0, stream>>>(feat2, w3h, b3, feat3, FRAMES);
    }

    // 4. bilinear query sampling
    sample_query_kernel<<<BB * NQ, CH, 0, stream>>>(feat3, qp, qh);

    // 5. fused correlation + softmax + soft-argmax + occlusion
    corr_softmax_kernel<<<BB * (NQ / 16) * TT, 128, 0, stream>>>(qh, feat3, out);
    (void)in_sizes; (void)n_in; (void)out_size; (void)ws_size;
}
